// GSSM_v2_v3_35605278884491
// MI455X (gfx1250) — matrix operand order repaired and checked by reading, not compiled, not run on hardware
//
#include <hip/hip_runtime.h>
#include <hip/hip_bf16.h>

// ---------------- problem constants ----------------
#define BB 4
#define LL 32768
#define NN (BB*LL)
#define DD 128
#define DI 256
#define DS 16
#define DTR 8
#define ZC 32
#define YC 512
#define XC 512
#define NCH 32           // scan chunks per sequence
#define CHUNK 1024       // L / NCH
#define TSZ 64           // LDS staging tile inside a chunk

typedef __attribute__((ext_vector_type(16))) _Float16 v16h;
typedef __attribute__((ext_vector_type(8)))  _Float16 h8;
typedef __attribute__((ext_vector_type(8)))  float    v8f;

// ---------------- WMMA tile loaders (CDNA5 f16 16x16x32 layouts) ----------------
// A: 16x32 (MxK). lane<16: row=lane, K={0..7,16..23}; lane>=16: row=lane-16, K={8..15,24..31}
__device__ __forceinline__ v16h ld_a16x32(const _Float16* base, int ld, int lane) {
  int row = lane & 15;
  int kb  = (lane >> 4) << 3;
  const _Float16* p = base + (size_t)row * ld + kb;
  h8 x0 = *(const h8*)(p);
  h8 x1 = *(const h8*)(p + 16);
  v16h r;
#pragma unroll
  for (int i = 0; i < 8; ++i) { r[i] = x0[i]; r[i + 8] = x1[i]; }
  return r;
}
// B: 32x16 (KxN), column n = weight row n. lane<16: col=lane, K={0..7,16..23}; lane>=16: col=lane-16, K={8..15,24..31}
__device__ __forceinline__ v16h ld_b32x16(const _Float16* wrow0, int ld, int lane) {
  int col = lane & 15;
  int kh  = (lane >> 4) << 3;
  const _Float16* p = wrow0 + (size_t)col * ld + kh;
  h8 x0 = *(const h8*)(p);
  h8 x1 = *(const h8*)(p + 16);
  v16h r;
#pragma unroll
  for (int i = 0; i < 8; ++i) { r[i] = x0[i]; r[i + 8] = x1[i]; }
  return r;
}

__device__ __forceinline__ float siluf(float v) { return v / (1.f + __expf(-v)); }

// ---------------- 1) keys = template[flat(coords)] ----------------
__global__ void k_keys(const int* __restrict__ coords, const int* __restrict__ tmpl,
                       int* __restrict__ keys) {
  int n = blockIdx.x * 256 + threadIdx.x;
  if (n >= NN) return;
  int z = coords[n * 4 + 1], y = coords[n * 4 + 2], x = coords[n * 4 + 3];
  keys[n] = tmpl[z * (YC * XC) + y * XC + x];
}

// ---------------- 2) weight conversion to f16 (x_proj zero-padded 40->48 rows) ------
__global__ void k_wconv(const float* __restrict__ Wi, const float* __restrict__ Wx,
                        const float* __restrict__ Wo,
                        _Float16* Wi16, _Float16* Wx16, _Float16* Wo16) {
  int i = blockIdx.x * 256 + threadIdx.x;             // 0 .. 65535
  if (i < 512 * 128) Wi16[i] = (_Float16)Wi[i];
  if (i < 48 * 256)  { int r = i >> 8; Wx16[i] = (_Float16)((r < 40) ? Wx[i] : 0.f); }
  if (i < 128 * 256) Wo16[i] = (_Float16)Wo[i];
}

// ---------------- 3) per-batch stable sort: 32768 u64 keys fully in 256KB LDS -------
extern __shared__ unsigned long long sm64[];
__global__ void k_sort(const int* __restrict__ keys, int* __restrict__ order) {
  unsigned b = blockIdx.x, tid = threadIdx.x;
  for (unsigned l = tid; l < (unsigned)LL; l += 1024u) {
    unsigned long long key = (unsigned long long)(unsigned)keys[b * LL + l];
    sm64[l] = (key << 15) | (unsigned long long)l;    // stable: tie-break by index
  }
  __syncthreads();
  for (unsigned k = 2; k <= (unsigned)LL; k <<= 1) {
    for (unsigned j = k >> 1; j > 0; j >>= 1) {
      for (unsigned i = tid; i < (unsigned)LL; i += 1024u) {
        unsigned ixj = i ^ j;
        if (ixj > i) {
          unsigned long long a = sm64[i], c = sm64[ixj];
          bool up = ((i & k) == 0u);
          if (up ? (a > c) : (a < c)) { sm64[i] = c; sm64[ixj] = a; }
        }
      }
      __syncthreads();
    }
  }
  for (unsigned l = tid; l < (unsigned)LL; l += 1024u)
    order[b * LL + l] = (int)(sm64[l] & 32767ull);
}

// ---------------- 4) gather + window-PE + input LayerNorm -> hn (f16) ----------------
__global__ void k_gather_ln(const float* __restrict__ feats, const int* __restrict__ coords,
                            const int* __restrict__ order,
                            const float* __restrict__ posW, const float* __restrict__ posb,
                            const float* __restrict__ lnw,  const float* __restrict__ lnb,
                            _Float16* __restrict__ hn) {
  __shared__ float red[128];
  int c = threadIdx.x;
  for (int t = 0; t < 4; ++t) {
    int g = blockIdx.x * 4 + t;                    // token in sorted domain
    int b = g >> 15;
    int orig = b * LL + order[g];
    float zc = (float)coords[orig * 4 + 1];
    float yc = (float)coords[orig * 4 + 2];
    float xc = (float)coords[orig * 4 + 3];
    float ycnt = 0.f, xcnt = 0.f, ycon = 0.f, xcon = 0.f, lys = 0.f, lxs = 0.f;
#pragma unroll
    for (int w = 0; w < 11; ++w) {                 // nwy = nwx = 11, step = 48, WIN = 64
      float s = 48.f * w;
      if (yc >= s && yc < s + 64.f) { ycnt += 1.f; ycon += w / 11.f; lys += (yc - s) / 64.f; }
      if (xc >= s && xc < s + 64.f) { xcnt += 1.f; xcon += w / 11.f; lxs += (xc - s) / 64.f; }
    }
    float pe0 = zc / 32.f, pe1 = xcon * ycnt, pe2 = ycon * xcnt, pe3 = lxs * ycnt, pe4 = lys * xcnt;
    float v = feats[(size_t)orig * DD + c]
            + pe0 * posW[c] + pe1 * posW[128 + c] + pe2 * posW[256 + c]
            + pe3 * posW[384 + c] + pe4 * posW[512 + c] + posb[c];
    red[c] = v; __syncthreads();
    for (int st = 64; st > 0; st >>= 1) { if (c < st) red[c] += red[c + st]; __syncthreads(); }
    float mean = red[0] / 128.f; __syncthreads();
    float d = v - mean;
    red[c] = d * d; __syncthreads();
    for (int st = 64; st > 0; st >>= 1) { if (c < st) red[c] += red[c + st]; __syncthreads(); }
    float var = red[0] / 128.f; __syncthreads();
    hn[(size_t)g * DD + c] = (_Float16)(d * rsqrtf(var + 1e-5f) * lnw[c] + lnb[c]);
  }
}

// ---------------- 5) in_proj GEMM (WMMA): hn(Nx128) x W^T -> xs(256) / silu(zg)(256) --
__global__ void k_gemm_in(const _Float16* __restrict__ hn, const _Float16* __restrict__ Wi16,
                          _Float16* __restrict__ xs, _Float16* __restrict__ zs) {
  int lane = threadIdx.x & 31;
  int m0 = blockIdx.x * 16;
  int c0 = blockIdx.y * 64;
  v8f acc[4] = {};
#pragma unroll
  for (int kt = 0; kt < 4; ++kt) {
    v16h a = ld_a16x32(hn + (size_t)m0 * 128 + kt * 32, 128, lane);
#pragma unroll
    for (int j = 0; j < 4; ++j) {
      v16h bt = ld_b32x16(Wi16 + (size_t)(c0 + j * 16) * 128 + kt * 32, 128, lane);
      acc[j] = __builtin_amdgcn_wmma_f32_16x16x32_f16(false, a, false, bt,
                                                      (short)0, acc[j], false, false);
    }
  }
  bool isz = (c0 >= 256);
  int n = lane & 15, mh = (lane >> 4) << 3;
#pragma unroll
  for (int j = 0; j < 4; ++j)
#pragma unroll
    for (int r = 0; r < 8; ++r) {
      int tok = m0 + mh + r, col = c0 + j * 16 + n;
      float v = acc[j][r];
      if (!isz) xs[(size_t)tok * DI + col] = (_Float16)v;
      else      zs[(size_t)tok * DI + (col - 256)] = (_Float16)siluf(v);
    }
}

// ---------------- 6) depthwise causal conv (4 taps) + SiLU ----------------
__global__ void k_conv(const _Float16* __restrict__ xs, const float* __restrict__ convw,
                       const float* __restrict__ convb, _Float16* __restrict__ xcv) {
  int c = threadIdx.x;
  int g = blockIdx.x;
  int l = g & (LL - 1);
  float acc = convb[c];
#pragma unroll
  for (int k = 0; k < 4; ++k) {
    int ls = l + k - 3;
    if (ls >= 0) acc += (float)xs[(size_t)(g + k - 3) * DI + c] * convw[k * DI + c];
  }
  xcv[(size_t)g * DI + c] = (_Float16)siluf(acc);
}

// ---------------- 7) x_proj GEMM (WMMA): xc(Nx256) x W^T -> dt_in/B/C --------------
__global__ void k_gemm_xdbl(const _Float16* __restrict__ xcv, const _Float16* __restrict__ Wx16,
                            float* __restrict__ dtin, float* __restrict__ Bm,
                            float* __restrict__ Cm) {
  __shared__ float tile[16][48];
  int lane = threadIdx.x & 31;
  int m0 = blockIdx.x * 16;
  v8f acc[3] = {};
#pragma unroll
  for (int kt = 0; kt < 8; ++kt) {
    v16h a = ld_a16x32(xcv + (size_t)m0 * 256 + kt * 32, 256, lane);
#pragma unroll
    for (int j = 0; j < 3; ++j) {
      v16h bt = ld_b32x16(Wx16 + (size_t)(j * 16) * 256 + kt * 32, 256, lane);
      acc[j] = __builtin_amdgcn_wmma_f32_16x16x32_f16(false, a, false, bt,
                                                      (short)0, acc[j], false, false);
    }
  }
  int n = lane & 15, mh = (lane >> 4) << 3;
#pragma unroll
  for (int j = 0; j < 3; ++j)
#pragma unroll
    for (int r = 0; r < 8; ++r) tile[mh + r][j * 16 + n] = acc[j][r];
  __syncthreads();
  for (int idx = lane; idx < 16 * 40; idx += 32) {
    int t = idx / 40, cc = idx % 40, tok = m0 + t;
    float v = tile[t][cc];
    if (cc < 8)       dtin[(size_t)tok * 8 + cc] = v;
    else if (cc < 24) Bm[(size_t)tok * 16 + (cc - 8)] = v;
    else              Cm[(size_t)tok * 16 + (cc - 24)] = v;
  }
}

// ---------------- 8) chunked parallel selective scan ----------------
__device__ __forceinline__ float softplusf(float x) {
  return (x > 20.f) ? x : log1pf(__expf(x));
}

// pass 1: per-chunk (prod dA, h_end | h0=0) ; dt computed in-register from dt_in (K=8)
__global__ void k_scan1(const _Float16* __restrict__ xcv, const float* __restrict__ dtin,
                        const float* __restrict__ Bm,
                        const float* __restrict__ dtW, const float* __restrict__ dtb,
                        const float* __restrict__ Alog,
                        float* __restrict__ Aprod, float* __restrict__ hend) {
  __shared__ float sdt[TSZ][8];
  __shared__ float sB[TSZ][16];
  int c = threadIdx.x, ch = blockIdx.x, b = blockIdx.y;
  int g0 = b * LL + ch * CHUNK;
  float w[8];
#pragma unroll
  for (int r = 0; r < 8; ++r) w[r] = dtW[c * 8 + r];
  float bias = dtb[c];
  float As[16], h[16], ap[16];
#pragma unroll
  for (int s = 0; s < 16; ++s) { As[s] = -__expf(Alog[c * 16 + s]); h[s] = 0.f; ap[s] = 1.f; }
  for (int tile = 0; tile < CHUNK / TSZ; ++tile) {
    int base = g0 + tile * TSZ;
    for (int i = c; i < TSZ * 8; i += 256)  sdt[i >> 3][i & 7]  = dtin[(size_t)(base + (i >> 3)) * 8 + (i & 7)];
    for (int i = c; i < TSZ * 16; i += 256) sB[i >> 4][i & 15]  = Bm[(size_t)(base + (i >> 4)) * 16 + (i & 15)];
    __syncthreads();
    for (int t = 0; t < TSZ; ++t) {
      float xacc = bias;
#pragma unroll
      for (int r = 0; r < 8; ++r) xacc += sdt[t][r] * w[r];
      float dt = softplusf(xacc);
      float xt = (float)xcv[(size_t)(base + t) * DI + c];
      float dx = dt * xt;
#pragma unroll
      for (int s = 0; s < 16; ++s) {
        float dA = __expf(dt * As[s]);
        ap[s] *= dA;
        h[s] = dA * h[s] + dx * sB[t][s];
      }
    }
    __syncthreads();
  }
  int bc = b * DI + c;
#pragma unroll
  for (int s = 0; s < 16; ++s) {
    size_t idx = ((size_t)ch * (BB * DI) + bc) * 16 + s;
    Aprod[idx] = ap[s];
    hend[idx] = h[s];
  }
}

// serial chunk composition: h_in per chunk for every (b, c, s)
__global__ void k_combine(const float* __restrict__ Aprod, const float* __restrict__ hend,
                          float* __restrict__ hin) {
  int t = blockIdx.x * 256 + threadIdx.x;          // bc*16 + s, 16384 total
  if (t >= BB * DI * 16) return;
  float h = 0.f;
  for (int ch = 0; ch < NCH; ++ch) {
    size_t idx = (size_t)ch * (BB * DI * 16) + t;
    hin[idx] = h;
    h = Aprod[idx] * h + hend[idx];
  }
}

// pass 2: replay chunk with correct h_in; fuse y + D*x then * silu(z); emit f16 ys
__global__ void k_scan2(const _Float16* __restrict__ xcv, const float* __restrict__ dtin,
                        const float* __restrict__ Bm, const float* __restrict__ Cm,
                        const float* __restrict__ dtW, const float* __restrict__ dtb,
                        const float* __restrict__ Alog, const float* __restrict__ Dp,
                        const _Float16* __restrict__ zs, const float* __restrict__ hin,
                        _Float16* __restrict__ ys) {
  __shared__ float sdt[TSZ][8];
  __shared__ float sB[TSZ][16];
  __shared__ float sC[TSZ][16];
  int c = threadIdx.x, ch = blockIdx.x, b = blockIdx.y;
  int g0 = b * LL + ch * CHUNK;
  int bc = b * DI + c;
  float w[8];
#pragma unroll
  for (int r = 0; r < 8; ++r) w[r] = dtW[c * 8 + r];
  float bias = dtb[c], dparam = Dp[c];
  float As[16], h[16];
#pragma unroll
  for (int s = 0; s < 16; ++s) {
    As[s] = -__expf(Alog[c * 16 + s]);
    h[s] = hin[((size_t)ch * (BB * DI) + bc) * 16 + s];
  }
  for (int tile = 0; tile < CHUNK / TSZ; ++tile) {
    int base = g0 + tile * TSZ;
    for (int i = c; i < TSZ * 8; i += 256)  sdt[i >> 3][i & 7] = dtin[(size_t)(base + (i >> 3)) * 8 + (i & 7)];
    for (int i = c; i < TSZ * 16; i += 256) {
      sB[i >> 4][i & 15] = Bm[(size_t)(base + (i >> 4)) * 16 + (i & 15)];
      sC[i >> 4][i & 15] = Cm[(size_t)(base + (i >> 4)) * 16 + (i & 15)];
    }
    __syncthreads();
    for (int t = 0; t < TSZ; ++t) {
      float xacc = bias;
#pragma unroll
      for (int r = 0; r < 8; ++r) xacc += sdt[t][r] * w[r];
      float dt = softplusf(xacc);
      float xt = (float)xcv[(size_t)(base + t) * DI + c];
      float dx = dt * xt;
      float y = 0.f;
#pragma unroll
      for (int s = 0; s < 16; ++s) {
        float dA = __expf(dt * As[s]);
        h[s] = dA * h[s] + dx * sB[t][s];
        y += h[s] * sC[t][s];
      }
      float zg = (float)zs[(size_t)(base + t) * DI + c];
      ys[(size_t)(base + t) * DI + c] = (_Float16)((y + dparam * xt) * zg);
    }
    __syncthreads();
  }
}

// ---------------- 9) out_proj GEMM (WMMA) + final LN + scatter-by-order -------------
__global__ void k_gemm_out(const _Float16* __restrict__ ys, const _Float16* __restrict__ Wo16,
                           const int* __restrict__ order,
                           const float* __restrict__ lnw, const float* __restrict__ lnb,
                           float* __restrict__ out) {
  __shared__ float tile[16][128];
  __shared__ float red[32], red2[32];
  int lane = threadIdx.x & 31;
  int m0 = blockIdx.x * 16;
  v8f acc[8] = {};
#pragma unroll
  for (int kt = 0; kt < 8; ++kt) {
    v16h a = ld_a16x32(ys + (size_t)m0 * 256 + kt * 32, 256, lane);
#pragma unroll
    for (int j = 0; j < 8; ++j) {
      v16h bt = ld_b32x16(Wo16 + (size_t)(j * 16) * 256 + kt * 32, 256, lane);
      acc[j] = __builtin_amdgcn_wmma_f32_16x16x32_f16(false, a, false, bt,
                                                      (short)0, acc[j], false, false);
    }
  }
  int n = lane & 15, mh = (lane >> 4) << 3;
#pragma unroll
  for (int j = 0; j < 8; ++j)
#pragma unroll
    for (int r = 0; r < 8; ++r) tile[mh + r][j * 16 + n] = acc[j][r];
  __syncthreads();
  for (int t = 0; t < 16; ++t) {
    float s = 0.f, s2 = 0.f;
#pragma unroll
    for (int q = 0; q < 4; ++q) { float v = tile[t][lane + 32 * q]; s += v; s2 += v * v; }
    red[lane] = s; red2[lane] = s2; __syncthreads();
    for (int st = 16; st > 0; st >>= 1) {
      if (lane < st) { red[lane] += red[lane + st]; red2[lane] += red2[lane + st]; }
      __syncthreads();
    }
    float mean = red[0] / 128.f;
    float var  = red2[0] / 128.f - mean * mean;
    float rstd = rsqrtf(var + 1e-5f);
    int g = m0 + t, b = g >> 15;
    size_t dst = ((size_t)b * LL + order[g]) * DD;   // out[b, order[b,l]] = o_sorted[b,l]
#pragma unroll
    for (int q = 0; q < 4; ++q) {
      int cc = lane + 32 * q;
      out[dst + cc] = (tile[t][cc] - mean) * rstd * lnw[cc] + lnb[cc];
    }
    __syncthreads();
  }
}

// ---------------- host launcher ----------------
extern "C" void kernel_launch(void* const* d_in, const int* in_sizes, int n_in,
                              void* d_out, int out_size, void* d_ws, size_t ws_size,
                              hipStream_t stream) {
  (void)in_sizes; (void)n_in; (void)out_size; (void)ws_size;
  const float* feats = (const float*)d_in[0];
  const int*   coords = (const int*)d_in[1];
  const int*   tmpl  = (const int*)d_in[2];
  const float* posW  = (const float*)d_in[3];
  const float* posb  = (const float*)d_in[4];
  const float* lniw  = (const float*)d_in[5];
  const float* lnib  = (const float*)d_in[6];
  const float* Wi    = (const float*)d_in[7];
  const float* convw = (const float*)d_in[8];
  const float* convb = (const float*)d_in[9];
  const float* Wx    = (const float*)d_in[10];
  const float* dtW   = (const float*)d_in[11];
  const float* dtb   = (const float*)d_in[12];
  const float* Alog  = (const float*)d_in[13];
  const float* Dp    = (const float*)d_in[14];
  const float* Wo    = (const float*)d_in[15];
  const float* slnw  = (const float*)d_in[16];
  const float* slnb  = (const float*)d_in[17];
  float* out = (float*)d_out;

  char* ws = (char*)d_ws;
  size_t off = 0;
  auto carve = [&](size_t bytes) -> void* {
    void* p = ws + off;
    off += (bytes + 255) & ~(size_t)255;
    return p;
  };
  int*      order = (int*)carve((size_t)NN * 4);
  int*      keys  = (int*)carve((size_t)NN * 4);
  _Float16* hn    = (_Float16*)carve((size_t)NN * DD * 2);
  _Float16* xs    = (_Float16*)carve((size_t)NN * DI * 2);
  _Float16* zs    = (_Float16*)carve((size_t)NN * DI * 2);
  _Float16* xcv   = (_Float16*)carve((size_t)NN * DI * 2);
  _Float16* ysb   = (_Float16*)carve((size_t)NN * DI * 2);
  float*    dtin  = (float*)carve((size_t)NN * 8 * 4);
  float*    Bmb   = (float*)carve((size_t)NN * 16 * 4);
  float*    Cmb   = (float*)carve((size_t)NN * 16 * 4);
  float*    Aprod = (float*)carve((size_t)NCH * BB * DI * 16 * 4);
  float*    hendb = (float*)carve((size_t)NCH * BB * DI * 16 * 4);
  float*    hinb  = (float*)carve((size_t)NCH * BB * DI * 16 * 4);
  _Float16* Wi16  = (_Float16*)carve((size_t)512 * 128 * 2);
  _Float16* Wx16  = (_Float16*)carve((size_t)48 * 256 * 2);
  _Float16* Wo16  = (_Float16*)carve((size_t)128 * 256 * 2);

  k_keys<<<NN / 256, 256, 0, stream>>>(coords, tmpl, keys);
  k_wconv<<<256, 256, 0, stream>>>(Wi, Wx, Wo, Wi16, Wx16, Wo16);
  k_sort<<<BB, 1024, (size_t)LL * sizeof(unsigned long long), stream>>>(keys, order);
  k_gather_ln<<<NN / 4, 128, 0, stream>>>(feats, coords, order, posW, posb, lniw, lnib, hn);
  k_gemm_in<<<dim3(NN / 16, 8), 32, 0, stream>>>(hn, Wi16, xs, zs);
  k_conv<<<NN, 256, 0, stream>>>(xs, convw, convb, xcv);
  k_gemm_xdbl<<<NN / 16, 32, 0, stream>>>(xcv, Wx16, dtin, Bmb, Cmb);
  k_scan1<<<dim3(NCH, BB), 256, 0, stream>>>(xcv, dtin, Bmb, dtW, dtb, Alog, Aprod, hendb);
  k_combine<<<(BB * DI * 16) / 256, 256, 0, stream>>>(Aprod, hendb, hinb);
  k_scan2<<<dim3(NCH, BB), 256, 0, stream>>>(xcv, dtin, Bmb, Cmb, dtW, dtb, Alog, Dp, zs,
                                             hinb, ysb);
  k_gemm_out<<<NN / 16, 32, 0, stream>>>(ysb, Wo16, order, slnw, slnb, out);
}
